// MultiheadSelfAttention_5729486373359
// MI455X (gfx1250) — matrix operand order repaired and checked by reading, not compiled, not run on hardware
//
#include <hip/hip_runtime.h>

// ---------------------------------------------------------------------------
// MI455X (gfx1250, wave32) multihead self-attention, bf16 WMMA + async-to-LDS.
// B=4, T=2048, D=2048, H=16, Hd=128.  V is stored pre-transposed by its GEMM
// so every WMMA B-fragment in attention is a contiguous 2x16B LDS read.
// ---------------------------------------------------------------------------

typedef __attribute__((ext_vector_type(16))) __bf16 bf16x16;
typedef __attribute__((ext_vector_type(8)))  __bf16 bf16x8;
typedef __attribute__((ext_vector_type(4)))  __bf16 bf16x4;
typedef __attribute__((ext_vector_type(8)))  float  f32x8;
typedef __attribute__((ext_vector_type(4)))  float  f32x4;
typedef __attribute__((ext_vector_type(4)))  int    i32x4;

#define B_   4
#define T_   2048
#define D_   2048
#define H_   16
#define HD_  128
#define BT_  (B_ * T_)
#define RMS_EPS 1.1920928955078125e-07f
#define Q_SCALE 0.08838834764831845f /* 1/sqrt(128) */

// ---- gfx1250 async global->LDS path (probe with __has_builtin) -------------
#if defined(__has_builtin)
#if __has_builtin(__builtin_amdgcn_global_load_async_to_lds_b128)
#define HAS_ASYNC_LDS 1
#endif
#if __has_builtin(__builtin_amdgcn_s_wait_asynccnt)
#define HAS_WAIT_ASYNC 1
#endif
#endif

// builtin signature: (v4i32 addrspace(1)*, v4i32 addrspace(3)*, imm off, imm cpol)
typedef __attribute__((address_space(1))) i32x4 gv4_t;
typedef __attribute__((address_space(3))) i32x4 lv4_t;

__device__ __forceinline__ void async_cp16(const __bf16* g, __bf16* l) {
#if defined(HAS_ASYNC_LDS)
  __builtin_amdgcn_global_load_async_to_lds_b128((gv4_t*)g, (lv4_t*)l, 0, 0);
#else
  *(bf16x8*)l = *(const bf16x8*)g;
#endif
}

template <int N>
__device__ __forceinline__ void wait_asynccnt() {
#if defined(HAS_WAIT_ASYNC)
  __builtin_amdgcn_s_wait_asynccnt(N);
#elif defined(HAS_ASYNC_LDS)
  asm volatile("s_wait_asynccnt %0" ::"n"(N) : "memory");
#endif
}

__device__ __forceinline__ f32x8 wmma_bf16(bf16x16 a, bf16x16 b, f32x8 c) {
  return __builtin_amdgcn_wmma_f32_16x16x32_bf16(
      /*neg_a=*/false, a, /*neg_b=*/false, b,
      /*c_mod=*/(short)0, c, /*reuse_a=*/false, /*reuse_b=*/false);
}

__device__ __forceinline__ f32x8 zero8() {
  f32x8 z = {0.f, 0.f, 0.f, 0.f, 0.f, 0.f, 0.f, 0.f};
  return z;
}

// A-matrix fragment (16x32 bf16, MxK) from [m][k] row-major LDS.
__device__ __forceinline__ bf16x16 frag_rowK(const __bf16* s, int ld, int m0, int k0) {
  const int lane = threadIdx.x & 31;
  const int m  = m0 + (lane & 15);
  const int kb = k0 + ((lane >> 4) << 3);
  bf16x8 lo = *(const bf16x8*)(s + m * ld + kb);
  bf16x8 hi = *(const bf16x8*)(s + m * ld + kb + 16);
  return __builtin_shufflevector(lo, hi, 0, 1, 2, 3, 4, 5, 6, 7,
                                         8, 9, 10, 11, 12, 13, 14, 15);
}

// B-matrix fragment (32x16 bf16, KxN) from [n][k]-contiguous LDS.
__device__ __forceinline__ bf16x16 frag_colK(const __bf16* s, int ld, int n0, int k0) {
  const int lane = threadIdx.x & 31;
  const int n  = n0 + (lane & 15);
  const int kb = k0 + ((lane >> 4) << 3);
  bf16x8 lo = *(const bf16x8*)(s + n * ld + kb);
  bf16x8 hi = *(const bf16x8*)(s + n * ld + kb + 16);
  return __builtin_shufflevector(lo, hi, 0, 1, 2, 3, 4, 5, 6, 7,
                                         8, 9, 10, 11, 12, 13, 14, 15);
}

// ---------------------------------------------------------------------------
// fp32 -> bf16 conversion
// ---------------------------------------------------------------------------
__global__ __launch_bounds__(256) void cvt4_kernel(const float* __restrict__ in,
                                                   __bf16* __restrict__ out, int n4) {
  int i = blockIdx.x * 256 + threadIdx.x;
  if (i < n4) {
    f32x4 v = ((const f32x4*)in)[i];
    bf16x4 o;
    o[0] = (__bf16)v[0]; o[1] = (__bf16)v[1];
    o[2] = (__bf16)v[2]; o[3] = (__bf16)v[3];
    ((bf16x4*)out)[i] = o;
  }
}

// ---------------------------------------------------------------------------
// GEMM: Y = A @ W^T + bias. Block 128x128, k-step 64, async double-buffered.
// 8 waves, wave tile 64x32 -> 16 WMMA per k-tile, one barrier per k-tile.
// MODE_Q/MODE_K: fused per-head RMSNorm, bf16 out (row-major).
// MODE_V: bias, bf16 out TRANSPOSED as [(b*H+h)*Hd + d][T].
// MODE_O: bias, fp32 out (row-major).
// ---------------------------------------------------------------------------
#define MODE_Q 0
#define MODE_K 1
#define MODE_V 2
#define MODE_O 3

template <int MODE>
__global__ __launch_bounds__(256) void gemm_kernel(
    const __bf16* __restrict__ A, const __bf16* __restrict__ W,
    const float* __restrict__ bias, const float* __restrict__ normw,
    float scale, void* __restrict__ out, int M, int N, int K) {
  __shared__ char smem[65536];
  // buf b: As[128][64] bf16 @ b*32768, Bs[128][64] bf16 @ b*32768+16384
  float* Cb = (float*)smem;  // epilogue overlay [128][128] fp32

  const int t    = threadIdx.x;
  const int lane = t & 31;
  const int wid  = t >> 5;
  const int wm   = wid >> 2;   // 0..1 -> 64 rows
  const int wn   = wid & 3;    // 0..3 -> 32 cols
  const int l15  = lane & 15;
  const int hi8  = (lane >> 4) << 3;
  const int m0   = blockIdx.y * 128;
  const int n0   = blockIdx.x * 128;

  auto stage = [&](int kt, int bsel) {
    const int k0 = kt << 6;
    __bf16* As = (__bf16*)(smem + bsel * 32768);
    __bf16* Bs = (__bf16*)(smem + bsel * 32768 + 16384);
#pragma unroll
    for (int j = 0; j < 4; ++j) {
      const int c  = t + (j << 8);           // 1024 16B chunks per tile
      const int r  = c >> 3;
      const int c8 = (c & 7) << 3;
      async_cp16(A + (size_t)(m0 + r) * K + k0 + c8, As + r * 64 + c8);
      async_cp16(W + (size_t)(n0 + r) * K + k0 + c8, Bs + r * 64 + c8);
    }
  };

  f32x8 acc[4][2];
#pragma unroll
  for (int mi = 0; mi < 4; ++mi) { acc[mi][0] = zero8(); acc[mi][1] = zero8(); }

  const int KT = K >> 6;
  stage(0, 0);
  for (int kt = 0; kt < KT; ++kt) {
    const int bsel = kt & 1;
    wait_asynccnt<0>();
    __syncthreads();  // tile kt visible to all; previous reads of other buf done
    if (kt + 1 < KT) stage(kt + 1, bsel ^ 1);
    if (kt + 2 < KT) {
      __builtin_prefetch(A + (size_t)(m0 + (t >> 3)) * K + (kt << 6) + 128, 0, 1);
      __builtin_prefetch(W + (size_t)(n0 + (t >> 3)) * K + (kt << 6) + 128, 0, 1);
    }
    const __bf16* As = (const __bf16*)(smem + bsel * 32768);
    const __bf16* Bs = (const __bf16*)(smem + bsel * 32768 + 16384);
#pragma unroll
    for (int ks = 0; ks < 2; ++ks) {
      bf16x16 bf0 = frag_colK(Bs, 64, wn * 32, ks * 32);
      bf16x16 bf1 = frag_colK(Bs, 64, wn * 32 + 16, ks * 32);
#pragma unroll
      for (int mi = 0; mi < 4; ++mi) {
        bf16x16 af = frag_rowK(As, 64, wm * 64 + mi * 16, ks * 32);
        acc[mi][0] = wmma_bf16(af, bf0, acc[mi][0]);
        acc[mi][1] = wmma_bf16(af, bf1, acc[mi][1]);
      }
    }
  }

  if (MODE == MODE_V || MODE == MODE_O) {
#pragma unroll
    for (int mi = 0; mi < 4; ++mi) {
#pragma unroll
      for (int ni = 0; ni < 2; ++ni) {
        const int n  = n0 + wn * 32 + ni * 16 + l15;
        const float bv = bias[n];
#pragma unroll
        for (int r = 0; r < 8; ++r) {
          const int m = m0 + wm * 64 + mi * 16 + hi8 + r;
          const float v = acc[mi][ni][r] + bv;
          if (MODE == MODE_O) {
            ((float*)out)[(size_t)m * N + n] = v;
          } else {
            // V^T store: m = b*T + tt, n = h*128 + d -> [(b*H+h)*Hd+d][T] @ tt
            const int bb = m >> 11, tt = m & (T_ - 1);
            const size_t row = ((size_t)bb * H_ + (n >> 7)) * HD_ + (n & 127);
            ((__bf16*)out)[row * T_ + tt] = (__bf16)v;
          }
        }
      }
    }
  } else {
    // Q/K: fp32 tile -> LDS, fused per-head RMSNorm (BN == head_dim == 128)
    __syncthreads();  // staging dead, overlay Cb
#pragma unroll
    for (int mi = 0; mi < 4; ++mi) {
#pragma unroll
      for (int ni = 0; ni < 2; ++ni) {
        const int nl = wn * 32 + ni * 16 + l15;
        const float bv = bias[n0 + nl];
#pragma unroll
        for (int r = 0; r < 8; ++r)
          Cb[(wm * 64 + mi * 16 + hi8 + r) * 128 + nl] = acc[mi][ni][r] + bv;
      }
    }
    __syncthreads();
    const int row  = t >> 1;
    const int half = t & 1;
    float ss = 0.f;
#pragma unroll 8
    for (int c = 0; c < 128; ++c) {
      const float v = Cb[row * 128 + c];
      ss += v * v;
    }
    const float rstd = rsqrtf(ss * (1.0f / 128.0f) + RMS_EPS) * scale;
    __bf16* O = (__bf16*)out;
#pragma unroll 8
    for (int c = half * 64; c < half * 64 + 64; ++c) {
      const float v = Cb[row * 128 + c] * rstd * normw[c];
      O[(size_t)(m0 + row) * N + n0 + c] = (__bf16)v;
    }
  }
}

// ---------------------------------------------------------------------------
// Flash-attention: grid (T/64, H, B). 64 q-rows, Hd=128, 64-key tiles.
// Async K/V prefetch (V pre-transposed in global); S/P overlay the K buffer.
// ---------------------------------------------------------------------------
__global__ __launch_bounds__(256) void attn_kernel(
    const __bf16* __restrict__ Q, const __bf16* __restrict__ Kc,
    const __bf16* __restrict__ Vt, __bf16* __restrict__ O) {
  __shared__ char smem[50176];
  __bf16* Qs  = (__bf16*)smem;             // [64][128] bf16
  __bf16* Ks  = (__bf16*)(smem + 16384);   // [64][128] bf16 (S/P overlay after use)
  float*  Sb  = (float*)(smem + 16384);    // [64][64] fp32 scores
  __bf16* Pb  = (__bf16*)(smem + 16384);   // [64][64] bf16 probs
  __bf16* Vs  = (__bf16*)(smem + 32768);   // [128][64] bf16 = V^T tile [d][kk]
  float*  mst = (float*)(smem + 49152);    // [64] running max
  float*  lst = mst + 64;                  // [64] running sum
  float*  alp = mst + 128;                 // [64] rescale factor

  const int t    = threadIdx.x;
  const int lane = t & 31;
  const int wid  = t >> 5;
  const int l15  = lane & 15;
  const int hi8  = (lane >> 4) << 3;
  const int wm   = wid & 1;    // 0..1 -> 32 rows
  const int wn   = wid >> 1;   // 0..3 -> 16 S-cols / 32 O-cols

  const int qt = blockIdx.x, h = blockIdx.y, bz = blockIdx.z;
  const size_t qbase = (size_t)bz * T_ + (size_t)qt * 64;
  const int coff = h * HD_;
  const size_t vrow0 = ((size_t)bz * H_ + h) * HD_;  // first row of V^T block

  auto stageQ = [&]() {
#pragma unroll
    for (int j = 0; j < 4; ++j) {
      const int c = t + (j << 8), r = c >> 4, c8 = (c & 15) << 3;
      async_cp16(Q + (qbase + r) * D_ + coff + c8, Qs + r * 128 + c8);
    }
  };
  auto stageKV = [&](int kt) {
    const size_t kb = (size_t)bz * T_ + (size_t)kt * 64;
#pragma unroll
    for (int j = 0; j < 4; ++j) {
      // K tile [64 kk][128 d]
      const int c = t + (j << 8), r = c >> 4, c8 = (c & 15) << 3;
      async_cp16(Kc + (kb + r) * D_ + coff + c8, Ks + r * 128 + c8);
      // V^T tile [128 d][64 kk]
      const int rv = c >> 3, cv8 = (c & 7) << 3;
      async_cp16(Vt + (vrow0 + rv) * T_ + kt * 64 + cv8, Vs + rv * 64 + cv8);
    }
  };

  if (t < 64) { mst[t] = -1e30f; lst[t] = 0.f; }
  stageQ();
  stageKV(0);

  f32x8 oacc[2][2];
#pragma unroll
  for (int mi = 0; mi < 2; ++mi) { oacc[mi][0] = zero8(); oacc[mi][1] = zero8(); }

  for (int kt = 0; kt < T_ / 64; ++kt) {
    wait_asynccnt<0>();
    __syncthreads();  // Q/K/V tiles resident

    // --- S = Q K^T : wave tile 32(q) x 16(kk), K-dim = Hd = 4 k-steps
    f32x8 sacc[2] = {zero8(), zero8()};
#pragma unroll
    for (int ks = 0; ks < 4; ++ks) {
      bf16x16 bk = frag_colK(Ks, 128, wn * 16, ks * 32);
#pragma unroll
      for (int mi = 0; mi < 2; ++mi) {
        bf16x16 aq = frag_rowK(Qs, 128, wm * 32 + mi * 16, ks * 32);
        sacc[mi] = wmma_bf16(aq, bk, sacc[mi]);
      }
    }
    __syncthreads();  // all K reads done; S/P may overlay
#pragma unroll
    for (int mi = 0; mi < 2; ++mi)
#pragma unroll
      for (int r = 0; r < 8; ++r)
        Sb[(wm * 32 + mi * 16 + hi8 + r) * 64 + wn * 16 + l15] = sacc[mi][r];
    __syncthreads();

    // --- online softmax: 2 threads/row (adjacent lanes), shfl pair-reduce
    const int row  = t >> 1;
    const int half = t & 1;
    float sv[32];
    float mn = 0.f, al = 0.f, tot = 0.f;
    if (t < 128) {
      float lmax = -1e30f;
#pragma unroll
      for (int i = 0; i < 32; ++i) {
        sv[i] = Sb[row * 64 + half * 32 + i];
        lmax = fmaxf(lmax, sv[i]);
      }
      const float rm = fmaxf(lmax, __shfl_xor(lmax, 1));
      const float mo = mst[row];
      mn = fmaxf(mo, rm);
      al = __expf(mo - mn);
      float ps = 0.f;
#pragma unroll
      for (int i = 0; i < 32; ++i) {
        const float e = __expf(sv[i] - mn);
        ps += e;
        sv[i] = e;
      }
      tot = ps + __shfl_xor(ps, 1);
    }
    __syncthreads();  // all S reads complete before P overwrites
    if (t < 128) {
#pragma unroll
      for (int j = 0; j < 4; ++j) {
        bf16x8 e;
#pragma unroll
        for (int i = 0; i < 8; ++i) e[i] = (__bf16)sv[j * 8 + i];
        *(bf16x8*)(Pb + row * 64 + half * 32 + j * 8) = e;
      }
      if (half == 0) {
        lst[row] = lst[row] * al + tot;
        mst[row] = mn;
        alp[row] = al;
      }
    }
    __syncthreads();

    // --- rescale O by alpha[row]
#pragma unroll
    for (int mi = 0; mi < 2; ++mi)
#pragma unroll
      for (int r = 0; r < 8; ++r) {
        const float a = alp[wm * 32 + mi * 16 + hi8 + r];
        oacc[mi][0][r] *= a;
        oacc[mi][1][r] *= a;
      }

    // --- O += P V : wave tile 32(q) x 32(d), K-dim = 64 keys = 2 k-steps
#pragma unroll
    for (int ks = 0; ks < 2; ++ks) {
      bf16x16 ap0 = frag_rowK(Pb, 64, wm * 32, ks * 32);
      bf16x16 ap1 = frag_rowK(Pb, 64, wm * 32 + 16, ks * 32);
#pragma unroll
      for (int ni = 0; ni < 2; ++ni) {
        bf16x16 bv = frag_colK(Vs, 64, wn * 32 + ni * 16, ks * 32);
        oacc[0][ni] = wmma_bf16(ap0, bv, oacc[0][ni]);
        oacc[1][ni] = wmma_bf16(ap1, bv, oacc[1][ni]);
      }
    }
    __syncthreads();  // P/V reads done -> safe to async-prefetch next K/V
    if (kt + 1 < T_ / 64) stageKV(kt + 1);
  }

  // --- finalize: O /= l, store bf16 at [b*T+q][h*128+d]
#pragma unroll
  for (int mi = 0; mi < 2; ++mi)
#pragma unroll
    for (int ni = 0; ni < 2; ++ni) {
      const int ncol = coff + wn * 32 + ni * 16 + l15;
#pragma unroll
      for (int r = 0; r < 8; ++r) {
        const int rl = wm * 32 + mi * 16 + hi8 + r;
        const float inv = 1.0f / lst[rl];
        O[(qbase + rl) * D_ + ncol] = (__bf16)(oacc[mi][ni][r] * inv);
      }
    }
}

// ---------------------------------------------------------------------------
// Host launch
// ---------------------------------------------------------------------------
extern "C" void kernel_launch(void* const* d_in, const int* in_sizes, int n_in,
                              void* d_out, int out_size, void* d_ws, size_t ws_size,
                              hipStream_t stream) {
  (void)in_sizes; (void)n_in; (void)out_size; (void)ws_size;
  const float* x   = (const float*)d_in[0];
  const float* Wq  = (const float*)d_in[1];
  const float* bq  = (const float*)d_in[2];
  const float* Wk  = (const float*)d_in[3];
  const float* bk  = (const float*)d_in[4];
  const float* Wv  = (const float*)d_in[5];
  const float* bv  = (const float*)d_in[6];
  const float* qnw = (const float*)d_in[7];
  const float* knw = (const float*)d_in[8];
  const float* Wo  = (const float*)d_in[9];
  const float* bo  = (const float*)d_in[10];

  char* ws = (char*)d_ws;
  const size_t ACT = (size_t)BT_ * D_ * 2;  // 32MB bf16 activation
  const size_t WSZ = (size_t)D_ * D_ * 2;   // 8MB bf16 weight
  __bf16* xb   = (__bf16*)(ws);
  __bf16* Wqb  = (__bf16*)(ws + ACT);
  __bf16* Wkb  = (__bf16*)(ws + ACT + WSZ);
  __bf16* Wvb  = (__bf16*)(ws + ACT + 2 * WSZ);
  __bf16* Wob  = (__bf16*)(ws + ACT + 3 * WSZ);
  __bf16* Qn   = (__bf16*)(ws + ACT + 4 * WSZ);
  __bf16* Kn   = (__bf16*)(ws + 2 * ACT + 4 * WSZ);
  __bf16* Vtb  = (__bf16*)(ws + 3 * ACT + 4 * WSZ);  // V^T [(b*H+h)*Hd+d][T]
  __bf16* attb = (__bf16*)(ws + 4 * ACT + 4 * WSZ);

  const int nx4 = BT_ * D_ / 4, nw4 = D_ * D_ / 4;
  cvt4_kernel<<<nx4 / 256, 256, 0, stream>>>(x, xb, nx4);
  cvt4_kernel<<<nw4 / 256, 256, 0, stream>>>(Wq, Wqb, nw4);
  cvt4_kernel<<<nw4 / 256, 256, 0, stream>>>(Wk, Wkb, nw4);
  cvt4_kernel<<<nw4 / 256, 256, 0, stream>>>(Wv, Wvb, nw4);
  cvt4_kernel<<<nw4 / 256, 256, 0, stream>>>(Wo, Wob, nw4);

  dim3 gg(D_ / 128, BT_ / 128);
  gemm_kernel<MODE_Q><<<gg, 256, 0, stream>>>(xb, Wqb, bq, qnw, Q_SCALE, Qn, BT_, D_, D_);
  gemm_kernel<MODE_K><<<gg, 256, 0, stream>>>(xb, Wkb, bk, knw, 1.0f, Kn, BT_, D_, D_);
  gemm_kernel<MODE_V><<<gg, 256, 0, stream>>>(xb, Wvb, bv, nullptr, 1.0f, Vtb, BT_, D_, D_);

  attn_kernel<<<dim3(T_ / 64, H_, B_), 256, 0, stream>>>(Qn, Kn, Vtb, attb);

  gemm_kernel<MODE_O><<<gg, 256, 0, stream>>>(attb, Wob, bo, nullptr, 1.0f, d_out, BT_, D_, D_);
}
